// Attention_48421461296026
// MI455X (gfx1250) — matrix operand order repaired and checked by reading, not compiled, not run on hardware
//
#include <hip/hip_runtime.h>
#include <hip/hip_bf16.h>

typedef __attribute__((ext_vector_type(16))) _Float16 v16h;
typedef __attribute__((ext_vector_type(8)))  float    v8f;

#define SEQ    2320
#define EMB    768
#define NH     12
#define HD     64
#define QKVLD  2304          // 3*EMB
#define ATTN_SCALE 0.125f    // 64^-0.5

// ---- fragment helpers (layouts per CDNA5 ISA §7.12.2) ----------
// A-matrix 16x32 f16: lanes 0-15 row=lane, K=[k..k+7, k+16..k+23];
//                     lanes16-31 row=lane-16, K=[k+8..k+15, k+24..k+31]
// B-matrix 32x16 f16 (stored as N x K row-major): lanes 0-15 col=lane K=[k..k+7, k+16..k+23],
//                     lanes 16-31 col=lane-16 K=[k+8..k+15, k+24..k+31]
static __device__ __forceinline__ v16h ld_frag_pair(const _Float16* p0, const _Float16* p1) {
  v16h r;
  reinterpret_cast<float4*>(&r)[0] = *reinterpret_cast<const float4*>(p0);
  reinterpret_cast<float4*>(&r)[1] = *reinterpret_cast<const float4*>(p1);
  return r;
}

static __device__ __forceinline__ v8f wmma16(v16h a, v16h b, v8f c) {
  return __builtin_amdgcn_wmma_f32_16x16x32_f16(false, a, false, b, (short)0, c, false, false);
}

// async global -> LDS copy, 16 bytes per lane (CDNA5 GLOBAL_LOAD_ASYNC_TO_LDS_B128,
// tracked by ASYNCcnt; LDS address from a VGPR, memaddr = VGPR pair)
static __device__ __forceinline__ void async_g2l_b128(unsigned lds_off, const void* gaddr) {
  asm volatile("global_load_async_to_lds_b128 %0, %1, off"
               :: "v"(lds_off), "v"((unsigned long long)(size_t)gaddr)
               : "memory");
}
static __device__ __forceinline__ void wait_asynccnt0() {
  asm volatile("s_wait_asynccnt 0" ::: "memory");
}

// ---------------------------------------------------------------------------
// f32 -> f16 convert
__global__ void __launch_bounds__(256)
cvt_f32_f16(const float* __restrict__ in, _Float16* __restrict__ out, int n) {
  int i = blockIdx.x * 256 + threadIdx.x;
  if (i < n) out[i] = (_Float16)in[i];
}

// ---------------------------------------------------------------------------
// C[M x N] = A[M x K] * W[N x K]^T + bias  (f16 in, f32 accum, f16 or f32 out)
// block = 256 threads = 8 waves; wave tile = 32 (M) x 64 (N)
__global__ void __launch_bounds__(256)
gemm_wmma(const _Float16* __restrict__ A, int lda,
          const _Float16* __restrict__ W, int ldw,
          const float* __restrict__ bias,
          int M, int N, int K,
          _Float16* __restrict__ outH, float* __restrict__ outF, int ldc)
{
  const int lane = threadIdx.x & 31;
  const int wave = threadIdx.x >> 5;
  const int hi   = lane >> 4;
  const int l15  = lane & 15;

  const int n0 = blockIdx.y * 512 + wave * 64;
  if (n0 >= N) return;                     // uniform per-wave exit (no barriers here)

  const int nMt = (M + 15) >> 4;
  const int mt0 = min(2 * (int)blockIdx.x,     nMt - 1);
  const int mt1 = min(2 * (int)blockIdx.x + 1, nMt - 1);   // dup tile writes same values
  const int m0 = mt0 * 16, m1 = mt1 * 16;

  v8f acc[2][4] = {};

  const _Float16* Arow0 = A + (size_t)(m0 + l15) * lda;
  const _Float16* Arow1 = A + (size_t)(m1 + l15) * lda;

  for (int kk = 0; kk < K; kk += 32) {
    const int ka = kk + hi * 8;
    v16h a0 = ld_frag_pair(Arow0 + ka, Arow0 + ka + 16);
    v16h a1 = ld_frag_pair(Arow1 + ka, Arow1 + ka + 16);
    // unconditional speculative prefetch of the next-next K block (branch-free;
    // overshoot past the row end is harmless for prefetch)
    __builtin_prefetch(Arow0 + ka + 64, 0, 3);
    __builtin_prefetch(Arow1 + ka + 64, 0, 3);
    #pragma unroll
    for (int t = 0; t < 4; t++) {
      const _Float16* wr = W + (size_t)(n0 + t * 16 + l15) * ldw + kk + hi * 8;
      v16h b = ld_frag_pair(wr, wr + 16);
      acc[0][t] = wmma16(a0, b, acc[0][t]);
      acc[1][t] = wmma16(a1, b, acc[1][t]);
    }
  }

  // C/D layout: VGPR r -> row r (lanes 0-15) / row r+8 (lanes 16-31), col = lane&15
  #pragma unroll
  for (int t = 0; t < 4; t++) {
    const int n = n0 + t * 16 + l15;
    const float bv = bias ? bias[n] : 0.0f;
    #pragma unroll
    for (int r = 0; r < 8; r++) {
      const int mm0 = m0 + r + hi * 8;
      const int mm1 = m1 + r + hi * 8;
      const float v0 = acc[0][t][r] + bv;
      const float v1 = acc[1][t][r] + bv;
      if (outH) {
        outH[(size_t)mm0 * ldc + n] = (_Float16)v0;
        outH[(size_t)mm1 * ldc + n] = (_Float16)v1;
      } else {
        outF[(size_t)mm0 * ldc + n] = v0;
        outF[(size_t)mm1 * ldc + n] = v1;
      }
    }
  }
}

// ---------------------------------------------------------------------------
// RoPE in-place on Q and K halves of qkv[SEQ][2304]; each thread owns the
// (d, d+32) rotation pair for one (token, head) to avoid the in-place race.
__global__ void __launch_bounds__(256)
rope_inplace(_Float16* __restrict__ qkv)
{
  const int total = 2304 * NH * 32;        // grid tokens * heads * 32 pairs
  const int idx = blockIdx.x * 256 + threadIdx.x;
  if (idx >= total) return;
  const int d2 = idx & 31;
  const int t  = idx >> 5;
  const int h  = t % NH;
  const int s  = t / NH;                   // token index in 48x48 grid
  const int gr = s / 48, gc = s % 48;

  // freq = 10000^{-(d2>>1)/16}
  const float freq = __expf(-(float)(d2 >> 1) * 0.5756462732485114f); // ln(1e4)/16
  const float a1 = (float)gr * freq;       // dims [0,32): row angle
  const float a2 = (float)gc * freq;       // dims [32,64): col angle
  float s1, c1, s2, c2;
  __sincosf(a1, &s1, &c1);
  __sincosf(a2, &s2, &c2);

  const size_t base = (size_t)(s + 16) * QKVLD + h * HD + d2; // skip 16 task tokens
  #pragma unroll
  for (int part = 0; part < 2; part++) {   // part 0 = Q, part 1 = K
    const size_t bb = base + (size_t)(part ? EMB : 0);
    const float x1 = (float)qkv[bb];
    const float x2 = (float)qkv[bb + 32];
    qkv[bb]      = (_Float16)(x1 * c1 - x2 * s1);
    qkv[bb + 32] = (_Float16)(x2 * c2 + x1 * s2);
  }
}

// ---------------------------------------------------------------------------
// Flash attention: grid (head, row-block of 64); 4 waves x 16 query rows.
// K tile staged once per block via async global->LDS; V transposed through LDS;
// P through per-wave LDS to convert C-layout -> A-fragment layout.
__global__ void __launch_bounds__(128)
attn_wmma(const _Float16* __restrict__ qkv,
          const unsigned char* __restrict__ mask,
          _Float16* __restrict__ ctx)
{
  const int lane = threadIdx.x & 31;
  const int wave = threadIdx.x >> 5;
  const int hi   = lane >> 4;
  const int l15  = lane & 15;
  const int h    = blockIdx.x;
  int q0 = blockIdx.y * 64 + wave * 16;
  q0 = min(q0, SEQ - 16);                 // clamped waves duplicate identical work

  const _Float16* Qb = qkv + h * HD;
  const _Float16* Kb = qkv + EMB + h * HD;
  const _Float16* Vb = qkv + 2 * EMB + h * HD;

  __shared__ __align__(16) _Float16 sK[32 * 64];       // K tile row-major [key][dim]
  __shared__ __align__(16) _Float16 sVt[64 * 32];      // V tile transposed [dim][key]
  __shared__ __align__(16) _Float16 sP[4][16 * 32];    // per-wave P tile

  // preload Q fragments (16 rows x 64 dims = two 16x32 A-frags)
  v16h aq[2];
  #pragma unroll
  for (int i = 0; i < 2; i++) {
    const _Float16* p = Qb + (size_t)(q0 + l15) * QKVLD + i * 32 + hi * 8;
    aq[i] = ld_frag_pair(p, p + 16);
  }

  v8f o[4] = {};
  float mrow[8], lrow[8];
  #pragma unroll
  for (int r = 0; r < 8; r++) { mrow[r] = -1e30f; lrow[r] = 0.0f; }

  const int vkey = threadIdx.x & 31;        // V-staging: key this thread handles
  const int vd0  = (threadIdx.x >> 5) * 16; // V-staging: first dim this thread handles
  const int akey = threadIdx.x >> 2;        // K-staging: key (0..31)
  const int aseg = threadIdx.x & 3;         // K-staging: 16B segment (0..3)

  for (int j0 = 0; j0 < SEQ; j0 += 32) {
    __syncthreads();                       // protect sK/sVt reads of previous iter
    {
      // ---- async-stage K tile: 32 keys x 64 dims, 2 x b128 per thread ----
      const int krow = min(j0 + akey, SEQ - 1);
      const _Float16* gk = Kb + (size_t)krow * QKVLD + aseg * 8;
      const unsigned lk = (unsigned)(size_t)(sK + akey * 64 + aseg * 8);
      async_g2l_b128(lk,      gk);
      async_g2l_b128(lk + 64, gk + 32);    // segments aseg and aseg+4

      // ---- stage V tile transposed (explicit; 2-byte scatter) ----
      const int vrow = min(j0 + vkey, SEQ - 1);
      const _Float16* vp = Vb + (size_t)vrow * QKVLD + vd0;
      float4 va = *reinterpret_cast<const float4*>(vp);
      float4 vb = *reinterpret_cast<const float4*>(vp + 8);
      const _Float16* pa = reinterpret_cast<const _Float16*>(&va);
      const _Float16* pb = reinterpret_cast<const _Float16*>(&vb);
      #pragma unroll
      for (int i = 0; i < 8; i++) {
        sVt[(vd0 + i) * 32 + vkey]     = pa[i];
        sVt[(vd0 + 8 + i) * 32 + vkey] = pb[i];
      }

      // branch-free prefetch of next iteration's K/V rows into near caches
      const int pkrow = min(j0 + 32 + vkey, SEQ - 1);
      __builtin_prefetch(Kb + (size_t)pkrow * QKVLD, 0, 3);
      __builtin_prefetch(Vb + (size_t)pkrow * QKVLD, 0, 3);
    }
    wait_asynccnt0();                      // our async K copies landed in LDS
    __syncthreads();                       // everyone's copies visible

    // S tile: 16 rows x 32 keys = two accumulators, two K-steps over head_dim
    v8f s0 = {}, s1 = {};
    #pragma unroll
    for (int i = 0; i < 2; i++) {
      const _Float16* k0p = sK + (l15)      * 64 + i * 32 + hi * 8;
      const _Float16* k1p = sK + (16 + l15) * 64 + i * 32 + hi * 8;
      v16h b0 = ld_frag_pair(k0p, k0p + 16);
      v16h b1 = ld_frag_pair(k1p, k1p + 16);
      s0 = wmma16(aq[i], b0, s0);
      s1 = wmma16(aq[i], b1, s1);
    }

    const int jc0 = j0 + l15, jc1 = j0 + 16 + l15;
    const float mb0 = (jc0 < SEQ && mask[jc0]) ? 0.0f : -1e9f;
    const float mb1 = (jc1 < SEQ && mask[jc1]) ? 0.0f : -1e9f;

    float v0[8], v1[8], rx[8];
    #pragma unroll
    for (int r = 0; r < 8; r++) {
      v0[r] = s0[r] * ATTN_SCALE + mb0;
      v1[r] = s1[r] * ATTN_SCALE + mb1;
      rx[r] = fmaxf(v0[r], v1[r]);
    }
    // row-wise max: reduce across the 16 lanes holding this row
    #pragma unroll
    for (int mx = 1; mx <= 8; mx <<= 1)
      #pragma unroll
      for (int r = 0; r < 8; r++)
        rx[r] = fmaxf(rx[r], __shfl_xor(rx[r], mx, 32));

    float corr[8], p0[8], p1[8], rs[8];
    #pragma unroll
    for (int r = 0; r < 8; r++) {
      const float mnew = fmaxf(mrow[r], rx[r]);
      corr[r] = __expf(mrow[r] - mnew);
      mrow[r] = mnew;
      p0[r] = __expf(v0[r] - mnew);
      p1[r] = __expf(v1[r] - mnew);
      rs[r] = p0[r] + p1[r];
    }
    #pragma unroll
    for (int mx = 1; mx <= 8; mx <<= 1)
      #pragma unroll
      for (int r = 0; r < 8; r++)
        rs[r] += __shfl_xor(rs[r], mx, 32);
    #pragma unroll
    for (int r = 0; r < 8; r++) lrow[r] = lrow[r] * corr[r] + rs[r];
    #pragma unroll
    for (int t = 0; t < 4; t++)
      #pragma unroll
      for (int r = 0; r < 8; r++)
        o[t][r] *= corr[r];

    // P (C-layout) -> per-wave LDS -> A-fragment layout
    _Float16* pbuf = &sP[wave][0];
    #pragma unroll
    for (int r = 0; r < 8; r++) {
      pbuf[(r + hi * 8) * 32 + l15]      = (_Float16)p0[r];
      pbuf[(r + hi * 8) * 32 + 16 + l15] = (_Float16)p1[r];
    }
    const _Float16* pr = pbuf + l15 * 32 + hi * 8;   // same-wave LDS: in-order
    v16h ap = ld_frag_pair(pr, pr + 16);

    // O += P @ V  (B-frags from transposed V tile in LDS)
    #pragma unroll
    for (int t = 0; t < 4; t++) {
      const _Float16* vr = sVt + (t * 16 + l15) * 32 + hi * 8;
      v16h bv = ld_frag_pair(vr, vr + 16);
      o[t] = wmma16(ap, bv, o[t]);
    }
  }

  #pragma unroll
  for (int r = 0; r < 8; r++) {
    const float inv = 1.0f / lrow[r];
    const int row = q0 + r + hi * 8;
    #pragma unroll
    for (int t = 0; t < 4; t++)
      ctx[(size_t)row * EMB + h * HD + t * 16 + l15] = (_Float16)(o[t][r] * inv);
  }
}

// ---------------------------------------------------------------------------
extern "C" void kernel_launch(void* const* d_in, const int* in_sizes, int n_in,
                              void* d_out, int out_size, void* d_ws, size_t ws_size,
                              hipStream_t stream)
{
  const float* x            = (const float*)d_in[0];
  const unsigned char* mask = (const unsigned char*)d_in[1];
  const float* Wqkv         = (const float*)d_in[2];
  const float* bqkv         = (const float*)d_in[3];
  const float* Wproj        = (const float*)d_in[4];
  const float* bproj        = (const float*)d_in[5];
  float* out = (float*)d_out;

  char* ws = (char*)d_ws;
  size_t off = 0;
  auto carve = [&](size_t bytes) -> void* {
    void* p = ws + off;
    off += (bytes + 255) & ~(size_t)255;
    return p;
  };
  _Float16* xh     = (_Float16*)carve((size_t)SEQ * EMB * 2);
  _Float16* wqkvh  = (_Float16*)carve((size_t)3 * EMB * EMB * 2);
  _Float16* wprojh = (_Float16*)carve((size_t)EMB * EMB * 2);
  _Float16* qkvh   = (_Float16*)carve((size_t)SEQ * 3 * EMB * 2);
  _Float16* ctxh   = (_Float16*)carve((size_t)SEQ * EMB * 2);

  auto cdiv = [](int a, int b) { return (a + b - 1) / b; };

  cvt_f32_f16<<<cdiv(SEQ * EMB, 256), 256, 0, stream>>>(x, xh, SEQ * EMB);
  cvt_f32_f16<<<cdiv(3 * EMB * EMB, 256), 256, 0, stream>>>(Wqkv, wqkvh, 3 * EMB * EMB);
  cvt_f32_f16<<<cdiv(EMB * EMB, 256), 256, 0, stream>>>(Wproj, wprojh, EMB * EMB);

  // qkv = x @ Wqkv^T + bqkv   (M=2320, N=2304, K=768) -> f16
  gemm_wmma<<<dim3(cdiv(cdiv(SEQ, 16), 2), cdiv(3 * EMB, 512)), 256, 0, stream>>>(
      xh, EMB, wqkvh, EMB, bqkv, SEQ, 3 * EMB, EMB, qkvh, nullptr, 3 * EMB);

  rope_inplace<<<cdiv(2304 * NH * 32, 256), 256, 0, stream>>>(qkvh);

  attn_wmma<<<dim3(NH, cdiv(SEQ, 64)), 128, 0, stream>>>(qkvh, mask, ctxh);

  // out = ctx @ Wproj^T + bproj (M=2320, N=768, K=768) -> f32
  gemm_wmma<<<dim3(cdiv(cdiv(SEQ, 16), 2), cdiv(EMB, 512)), 256, 0, stream>>>(
      ctxh, EMB, wprojh, EMB, bproj, SEQ, EMB, EMB, nullptr, out, EMB);
}
